// PatchEmbed_46677704573735
// MI455X (gfx1250) — compile-verified, matrix operand order checked by reading, not run on hardware
//
#include <hip/hip_runtime.h>
#include <hip/hip_bf16.h>

// ---------------------------------------------------------------------------
// PatchEmbed for MI455X (gfx1250): FPS + KNN + fused PointNet MLP with WMMA.
// All GEMMs run as V_WMMA_F32_16X16X32_F16 (f16 in, f32 accumulate).
// Per-patch intermediates live entirely in the 320KB WGP LDS.
// ---------------------------------------------------------------------------

typedef _Float16 h8  __attribute__((ext_vector_type(8)));
typedef _Float16 h16 __attribute__((ext_vector_type(16)));
typedef float    f8  __attribute__((ext_vector_type(8)));

#define BATCH   4
#define NPTS    32768
#define SPATCH  512
#define KNBR    64
#define CFEAT   6
#define CIN     9
#define BN_EPS  1e-5f

// ---------------------------------------------------------------------------
// WMMA helpers
// ---------------------------------------------------------------------------
__device__ __forceinline__ f8 wmma_f16(h16 a, h16 b, f8 c) {
    // 8 args: (neg_a, A, neg_b, B, c_mod, C, reuse_a, reuse_b)
    return __builtin_amdgcn_wmma_f32_16x16x32_f16(false, a, false, b,
                                                  (short)0, c, false, false);
}

// Load a 16x32 fragment (A-layout; B uses same layout on W^T stored [N][K]).
// 16-bit A 16x32: lanes 0-15 -> K {0..7,16..23}, lanes 16-31 -> K {8..15,24..31}
__device__ __forceinline__ h16 load_frag(const _Float16* tile, int ld, int lane) {
    const int row = lane & 15;
    const int kb  = (lane & 16) ? 8 : 0;
    const _Float16* p = tile + (size_t)row * ld + kb;
    union { h16 v; h8 h[2]; } u;
    u.h[0] = *(const h8*)(p);
    u.h[1] = *(const h8*)(p + 16);
    return u.v;
}

// ---------------------------------------------------------------------------
// Kernel 1: fold BN into weights, convert to f16, transpose to [N][K]
// ---------------------------------------------------------------------------
__global__ void prep_kernel(const float* __restrict__ W1, const float* __restrict__ b1,
                            const float* __restrict__ g1, const float* __restrict__ be1,
                            const float* __restrict__ m1, const float* __restrict__ v1,
                            const float* __restrict__ W2, const float* __restrict__ b2,
                            const float* __restrict__ W3, const float* __restrict__ b3,
                            const float* __restrict__ g3, const float* __restrict__ be3,
                            const float* __restrict__ m3, const float* __restrict__ v3,
                            const float* __restrict__ W4, const float* __restrict__ b4,
                            float* W1f, float* b1f,
                            _Float16* W2t, float* b2f,
                            _Float16* W3t, float* b3f,
                            _Float16* W4t, float* b4f)
{
    int i = blockIdx.x * blockDim.x + threadIdx.x;
    if (i < 1024 * 1024) {                        // W3t[n][k] = W3[k][n] * s3[n]
        int n = i >> 10, k = i & 1023;
        float s = g3[n] * rsqrtf(v3[n] + BN_EPS);
        W3t[i] = (_Float16)(W3[k * 1024 + n] * s);
    }
    if (i < 256 * 1024) {                         // W4t[n][k] = W4[k][n]
        int n = i >> 10, k = i & 1023;
        W4t[i] = (_Float16)(W4[k * 256 + n]);
    }
    if (i < 512 * 128) {                          // W2t[n][k] = W2[k][n]
        int n = i >> 7, k = i & 127;
        W2t[i] = (_Float16)(W2[k * 512 + n]);
    }
    if (i < CIN * 128) {                          // W1f[r][c] = W1[r][c]*s1[c] (f32)
        int c = i & 127;
        float s = g1[c] * rsqrtf(v1[c] + BN_EPS);
        W1f[i] = W1[i] * s;
    }
    if (i < 1024) { float s = g3[i] * rsqrtf(v3[i] + BN_EPS); b3f[i] = (b3[i] - m3[i]) * s + be3[i]; }
    if (i < 512)  b2f[i] = b2[i];
    if (i < 256)  b4f[i] = b4[i];
    if (i < 128)  { float s = g1[i] * rsqrtf(v1[i] + BN_EPS); b1f[i] = (b1[i] - m1[i]) * s + be1[i]; }
}

// ---------------------------------------------------------------------------
// Kernel 2: farthest point sampling. One 1024-thread workgroup per batch.
// Per-thread running min-distances stay in VGPRs (32 each); block argmax via
// wave shfl_xor then a single-wave LDS reduction.
// ---------------------------------------------------------------------------
#define FPS_T 1024
#define FPS_PPT (NPTS / FPS_T)   // 32

__global__ __launch_bounds__(FPS_T) void fps_kernel(
    const float* __restrict__ coords, float* __restrict__ centers,
    int* __restrict__ fps_idx)
{
    const int b = blockIdx.x;
    const float* C = coords + (size_t)b * NPTS * 3;
    const int t = threadIdx.x;

    __shared__ float lastc[3];
    __shared__ float swv[32];
    __shared__ int   swi[32];

    float dist[FPS_PPT];
#pragma unroll
    for (int i = 0; i < FPS_PPT; i++) dist[i] = 1e10f;

    if (t == 0) {
        lastc[0] = C[0]; lastc[1] = C[1]; lastc[2] = C[2];
        fps_idx[b * SPATCH] = 0;
        centers[(size_t)(b * SPATCH) * 3 + 0] = C[0];
        centers[(size_t)(b * SPATCH) * 3 + 1] = C[1];
        centers[(size_t)(b * SPATCH) * 3 + 2] = C[2];
    }
    __syncthreads();

    for (int s = 1; s < SPATCH; s++) {
        const float lx = lastc[0], ly = lastc[1], lz = lastc[2];
        float bv = -1.0f; int bi = 0;
#pragma unroll
        for (int i = 0; i < FPS_PPT; i++) {
            const int p = t + i * FPS_T;                 // coalesced
            float dx = C[p * 3 + 0] - lx;
            float dy = C[p * 3 + 1] - ly;
            float dz = C[p * 3 + 2] - lz;
            float d = dx * dx + dy * dy + dz * dz;
            dist[i] = fminf(dist[i], d);
            if (dist[i] > bv) { bv = dist[i]; bi = p; }
        }
        // wave32 reduction (max value, tie -> lower index)
#pragma unroll
        for (int o = 16; o > 0; o >>= 1) {
            float ov = __shfl_xor(bv, o);
            int   oi = __shfl_xor(bi, o);
            if (ov > bv || (ov == bv && oi < bi)) { bv = ov; bi = oi; }
        }
        if ((t & 31) == 0) { swv[t >> 5] = bv; swi[t >> 5] = bi; }
        __syncthreads();
        if (t < 32) {
            bv = swv[t]; bi = swi[t];
#pragma unroll
            for (int o = 16; o > 0; o >>= 1) {
                float ov = __shfl_xor(bv, o);
                int   oi = __shfl_xor(bi, o);
                if (ov > bv || (ov == bv && oi < bi)) { bv = ov; bi = oi; }
            }
            if (t == 0) {
                fps_idx[b * SPATCH + s] = bi;
                float x = C[bi * 3 + 0], y = C[bi * 3 + 1], z = C[bi * 3 + 2];
                lastc[0] = x; lastc[1] = y; lastc[2] = z;
                size_t o = (size_t)(b * SPATCH + s) * 3;
                centers[o + 0] = x; centers[o + 1] = y; centers[o + 2] = z;
            }
        }
        __syncthreads();
    }
}

// ---------------------------------------------------------------------------
// Kernel 3: KNN top-64. One 256-thread workgroup per center.
// d2[N] lives in LDS (128KB). 64 incremental argmin rounds: only the winning
// thread rescans its strided slice, so each round is just a block reduction.
// ---------------------------------------------------------------------------
#define KNN_T 256
#define KNN_PPT (NPTS / KNN_T)   // 128

__global__ __launch_bounds__(KNN_T) void knn_kernel(
    const float* __restrict__ coords, const float* __restrict__ centers,
    int* __restrict__ knn_idx)
{
    extern __shared__ unsigned char smem_u8[];
    float* d2 = (float*)smem_u8;                // NPTS floats

    const int sid = blockIdx.x;
    const int b = sid / SPATCH;
    const float* C = coords + (size_t)b * NPTS * 3;
    const float cx = centers[(size_t)sid * 3 + 0];
    const float cy = centers[(size_t)sid * 3 + 1];
    const float cz = centers[(size_t)sid * 3 + 2];
    const int t = threadIdx.x;

    __shared__ float swv[8];
    __shared__ int   swi[8];
    __shared__ int   winner;

    float lv = 1e30f; int li = -1;
    for (int i = 0; i < KNN_PPT; i++) {
        const int p = t + i * KNN_T;                   // coalesced + strided slice
        float dx = C[p * 3 + 0] - cx;
        float dy = C[p * 3 + 1] - cy;
        float dz = C[p * 3 + 2] - cz;
        float d = dx * dx + dy * dy + dz * dz;
        d2[p] = d;
        if (d < lv || (d == lv && p < li)) { lv = d; li = p; }
    }
    __syncthreads();

    for (int r = 0; r < KNBR; r++) {
        float bv = lv; int bi = li;
#pragma unroll
        for (int o = 16; o > 0; o >>= 1) {
            float ov = __shfl_xor(bv, o);
            int   oi = __shfl_xor(bi, o);
            if (ov < bv || (ov == bv && oi < bi)) { bv = ov; bi = oi; }
        }
        if ((t & 31) == 0) { swv[t >> 5] = bv; swi[t >> 5] = bi; }
        __syncthreads();
        if (t == 0) {
            float wv = swv[0]; int wi = swi[0];
#pragma unroll
            for (int w = 1; w < 8; w++)
                if (swv[w] < wv || (swv[w] == wv && swi[w] < wi)) { wv = swv[w]; wi = swi[w]; }
            winner = wi;
            knn_idx[(size_t)sid * KNBR + r] = wi;
        }
        __syncthreads();
        const int wi = winner;
        if ((wi & (KNN_T - 1)) == t) {                 // owner rescans its slice
            d2[wi] = 1e30f;
            lv = 1e30f; li = -1;
            for (int i = 0; i < KNN_PPT; i++) {
                const int p = t + i * KNN_T;
                float d = d2[p];
                if (d < lv) { lv = d; li = p; }
            }
        }
        __syncthreads();
    }
}

// ---------------------------------------------------------------------------
// Kernel 4: fused per-patch PointNet MLP with WMMA.
// One 512-thread workgroup (16 wave32 -> 4 waves/SIMD) per patch.
// LDS: act[64][1024] f16 + z[64][1024] f16 + gather/gmax = ~261KB (<320KB).
// Weight streams are prefetched ahead with global_prefetch_b8.
// ---------------------------------------------------------------------------
#define MLP_T 512

__global__ __launch_bounds__(MLP_T) void patch_mlp_kernel(
    const float* __restrict__ coords, const float* __restrict__ features,
    const float* __restrict__ centers, const int* __restrict__ knn_idx,
    const float* __restrict__ W1f, const float* __restrict__ b1f,
    const _Float16* __restrict__ W2t, const float* __restrict__ b2f,
    const _Float16* __restrict__ W3t, const float* __restrict__ b3f,
    const _Float16* __restrict__ W4t, const float* __restrict__ b4f,
    float* __restrict__ out)
{
    extern __shared__ unsigned char smem_u8[];
    _Float16* act = (_Float16*)smem_u8;          // [64][1024]
    _Float16* z   = act + 64 * 1024;             // [64][1024]
    float* pin    = (float*)(z + 64 * 1024);     // [64][12]
    float* gmax   = pin + 64 * 12;               // [512]

    const int sid = blockIdx.x;
    const int b = sid / SPATCH;
    const int t = threadIdx.x;
    const int lane = t & 31;
    const int w = t >> 5;                        // 16 waves

    // ---- gather 64 neighbors: centered coords + features -> pin[64][9] ----
    if (t < KNBR) {
        const int idx = knn_idx[(size_t)sid * KNBR + t];
        const float* P = coords   + ((size_t)b * NPTS + idx) * 3;
        const float* F = features + ((size_t)b * NPTS + idx) * CFEAT;
        const float cx = centers[(size_t)sid * 3 + 0];
        const float cy = centers[(size_t)sid * 3 + 1];
        const float cz = centers[(size_t)sid * 3 + 2];
        pin[t * 12 + 0] = P[0] - cx;
        pin[t * 12 + 1] = P[1] - cy;
        pin[t * 12 + 2] = P[2] - cz;
#pragma unroll
        for (int j = 0; j < CFEAT; j++) pin[t * 12 + 3 + j] = F[j];
    }
    // warm the L2->WGP path for this wave's first stage-C weight rows
    __builtin_prefetch(W3t + (size_t)(w * 4) * 16 * 1024 + (lane & 15) * 1024, 0, 1);
    __syncthreads();

    // ---- stage A (scalar, tiny): [64,9]@W1f + b1f, ReLU -> z[:,0:128] f16 ----
    {
        const int row = t >> 3;                  // 0..63
        const int c0  = (t & 7) * 16;            // 8 threads per row
        float in[CIN];
#pragma unroll
        for (int i = 0; i < CIN; i++) in[i] = pin[row * 12 + i];
        for (int cc = 0; cc < 16; cc++) {
            const int c = c0 + cc;
            float a = b1f[c];
#pragma unroll
            for (int i = 0; i < CIN; i++) a += in[i] * W1f[i * 128 + c];
            z[row * 1024 + c] = (_Float16)fmaxf(a, 0.0f);
        }
    }
    __syncthreads();

    // ---- stage B (WMMA): [64,128]@W2t -> act[:,512:1024], column max -> gmax ----
    for (int j = 0; j < 2; j++) {
        const int nt = w * 2 + j;                    // 0..31 (N=512)
        f8 acc[4] = {};
        for (int k = 0; k < 4; k++) {
            h16 bf = load_frag(W2t + (size_t)nt * 16 * 128 + k * 32, 128, lane);
#pragma unroll
            for (int m = 0; m < 4; m++) {
                h16 af = load_frag(z + m * 16 * 1024 + k * 32, 1024, lane);
                acc[m] = wmma_f16(af, bf, acc[m]);
            }
        }
        const int col = nt * 16 + (lane & 15);
        const float bias = b2f[col];
        float cmax = -1e30f;
#pragma unroll
        for (int m = 0; m < 4; m++) {
            const int rbase = ((lane >> 4) << 3) + m * 16;
#pragma unroll
            for (int r = 0; r < 8; r++) {
                float v = acc[m][r] + bias;
                act[(rbase + r) * 1024 + 512 + col] = (_Float16)v;
                cmax = fmaxf(cmax, v);
            }
        }
        cmax = fmaxf(cmax, __shfl_xor(cmax, 16));
        if (lane < 16) gmax[col] = cmax;
    }
    __syncthreads();

    // ---- broadcast gmax into act[:,0:512] (the concat's first half) ----
    {
        const int row = t >> 3;
        const int c0  = (t & 7) * 64;
        for (int cc = 0; cc < 64; cc++)
            act[row * 1024 + c0 + cc] = (_Float16)gmax[c0 + cc];
    }
    __syncthreads();

    // ---- stage C (WMMA): [64,1024]@W3t + b3f, ReLU -> z (f16) ----
    for (int j = 0; j < 4; j++) {
        const int nt = w * 4 + j;                    // 0..63 (N=1024)
        // prefetch next tile's weight row for this lane (global_prefetch_b8)
        if (j < 3)
            __builtin_prefetch(W3t + (size_t)(nt + 1) * 16 * 1024 + (lane & 15) * 1024, 0, 1);
        f8 acc[4] = {};
        for (int k = 0; k < 32; k++) {
            h16 bf = load_frag(W3t + (size_t)nt * 16 * 1024 + k * 32, 1024, lane);
#pragma unroll
            for (int m = 0; m < 4; m++) {
                h16 af = load_frag(act + m * 16 * 1024 + k * 32, 1024, lane);
                acc[m] = wmma_f16(af, bf, acc[m]);
            }
        }
        const int col = nt * 16 + (lane & 15);
        const float bias = b3f[col];
#pragma unroll
        for (int m = 0; m < 4; m++) {
            const int rbase = ((lane >> 4) << 3) + m * 16;
#pragma unroll
            for (int r = 0; r < 8; r++)
                z[(rbase + r) * 1024 + col] = (_Float16)fmaxf(acc[m][r] + bias, 0.0f);
        }
    }
    __syncthreads();

    // ---- stage D (WMMA): [64,1024]@W4t, max over the 64 points -> out ----
    {
        const int nt = w;                            // 0..15 (N=256)
        __builtin_prefetch(W4t + (size_t)nt * 16 * 1024 + (lane & 15) * 1024, 0, 1);
        f8 acc[4] = {};
        for (int k = 0; k < 32; k++) {
            h16 bf = load_frag(W4t + (size_t)nt * 16 * 1024 + k * 32, 1024, lane);
#pragma unroll
            for (int m = 0; m < 4; m++) {
                h16 af = load_frag(z + m * 16 * 1024 + k * 32, 1024, lane);
                acc[m] = wmma_f16(af, bf, acc[m]);
            }
        }
        float cmax = -1e30f;
#pragma unroll
        for (int m = 0; m < 4; m++)
#pragma unroll
            for (int r = 0; r < 8; r++) cmax = fmaxf(cmax, acc[m][r]);
        cmax = fmaxf(cmax, __shfl_xor(cmax, 16));    // fold the two M halves
        const int col = nt * 16 + (lane & 15);
        if (lane < 16)
            out[(size_t)sid * 256 + col] = cmax + b4f[col];   // bias after max (uniform per col)
    }
}

// ---------------------------------------------------------------------------
extern "C" void kernel_launch(void* const* d_in, const int* in_sizes, int n_in,
                              void* d_out, int out_size, void* d_ws, size_t ws_size,
                              hipStream_t stream) {
    const float* coords   = (const float*)d_in[0];
    const float* features = (const float*)d_in[1];
    const float* W1 = (const float*)d_in[2];
    const float* b1 = (const float*)d_in[3];
    const float* g1 = (const float*)d_in[4];
    const float* be1 = (const float*)d_in[5];
    const float* m1 = (const float*)d_in[6];
    const float* v1 = (const float*)d_in[7];
    const float* W2 = (const float*)d_in[8];
    const float* b2 = (const float*)d_in[9];
    const float* W3 = (const float*)d_in[10];
    const float* b3 = (const float*)d_in[11];
    const float* g3 = (const float*)d_in[12];
    const float* be3 = (const float*)d_in[13];
    const float* m3 = (const float*)d_in[14];
    const float* v3 = (const float*)d_in[15];
    const float* W4 = (const float*)d_in[16];
    const float* b4 = (const float*)d_in[17];
    float* out = (float*)d_out;

    char* ws = (char*)d_ws;
    size_t off = 0;
    auto take = [&](size_t bytes) -> char* {
        off = (off + 255) & ~(size_t)255;
        char* p = ws + off;
        off += bytes;
        return p;
    };
    float* centers = (float*)take((size_t)BATCH * SPATCH * 3 * sizeof(float));
    int*   fpsidx  = (int*)  take((size_t)BATCH * SPATCH * sizeof(int));
    int*   knn     = (int*)  take((size_t)BATCH * SPATCH * KNBR * sizeof(int));
    float* W1f = (float*)take(CIN * 128 * sizeof(float));
    float* b1f = (float*)take(128 * sizeof(float));
    _Float16* W2t = (_Float16*)take((size_t)512 * 128 * sizeof(_Float16));
    float* b2f = (float*)take(512 * sizeof(float));
    _Float16* W3t = (_Float16*)take((size_t)1024 * 1024 * sizeof(_Float16));
    float* b3f = (float*)take(1024 * sizeof(float));
    _Float16* W4t = (_Float16*)take((size_t)256 * 1024 * sizeof(_Float16));
    float* b4f = (float*)take(256 * sizeof(float));
    (void)ws_size; (void)n_in; (void)in_sizes; (void)out_size;

    // 1) fold BN, cast weights to f16, transpose to [N][K]
    prep_kernel<<<(1024 * 1024 + 255) / 256, 256, 0, stream>>>(
        W1, b1, g1, be1, m1, v1, W2, b2, W3, b3, g3, be3, m3, v3, W4, b4,
        W1f, b1f, W2t, b2f, W3t, b3f, W4t, b4f);

    // 2) farthest point sampling (one WGP per batch)
    fps_kernel<<<BATCH, FPS_T, 0, stream>>>(coords, centers, fpsidx);

    // 3) KNN top-64 (d2 in 128KB LDS)
    knn_kernel<<<BATCH * SPATCH, KNN_T, NPTS * sizeof(float), stream>>>(
        coords, centers, knn);

    // 4) fused per-patch MLP with WMMA (256KB f16 activations in LDS)
    const size_t smem = (size_t)2 * 64 * 1024 * sizeof(_Float16)   // act + z
                      + 64 * 12 * sizeof(float)                    // gathered input
                      + 512 * sizeof(float);                       // gmax
    patch_mlp_kernel<<<BATCH * SPATCH, MLP_T, smem, stream>>>(
        coords, features, centers, knn,
        W1f, b1f, W2t, b2f, W3t, b3f, W4t, b4f, out);
}
